// MultiHeadAttention__44014824849983
// MI455X (gfx1250) — matrix operand order repaired and checked by reading, not compiled, not run on hardware
//
#include <hip/hip_runtime.h>

typedef __attribute__((ext_vector_type(16))) __bf16 v16bf;
typedef __attribute__((ext_vector_type(8)))  float  v8f;
typedef __attribute__((ext_vector_type(4)))  unsigned u32x4;
typedef __attribute__((ext_vector_type(8)))  int      i32x8;
typedef __attribute__((ext_vector_type(4)))  int      i32x4;

#define B_   4
#define N_   2048
#define M_   2048
#define D_   1024
#define H_   16
#define HS_  64
#define OUT_ 1024

// ---- per-16-lane-half reductions (rows of a C tile live in one half) ----
static __device__ __forceinline__ float half_max(float v) {
  v = fmaxf(v, __shfl_xor(v, 1, 32));
  v = fmaxf(v, __shfl_xor(v, 2, 32));
  v = fmaxf(v, __shfl_xor(v, 4, 32));
  v = fmaxf(v, __shfl_xor(v, 8, 32));
  return v;
}
static __device__ __forceinline__ float half_sum(float v) {
  v += __shfl_xor(v, 1, 32);
  v += __shfl_xor(v, 2, 32);
  v += __shfl_xor(v, 4, 32);
  v += __shfl_xor(v, 8, 32);
  return v;
}

static __device__ __forceinline__ void wait_async0() {
#if __has_builtin(__builtin_amdgcn_s_wait_asynccnt)
  __builtin_amdgcn_s_wait_asynccnt(0);
#else
  asm volatile("s_wait_asynccnt 0x0" ::: "memory");
#endif
}
static __device__ __forceinline__ void wait_tensor0() {
#if __has_builtin(__builtin_amdgcn_s_wait_tensorcnt)
  __builtin_amdgcn_s_wait_tensorcnt(0);
#else
  asm volatile("s_wait_tensorcnt 0x0" ::: "memory");
#endif
}

// ---- TDM: move one 32-row x 128-byte 2D tile (bf16 32x64) global -> LDS ----
// D# per CDNA5 ISA ch.8: data_size=8B units, tensor_dim0=16u, stride0=16u,
// tile_dim0=16u, tile_dim1=32 rows, global_addr = tile start, type=2.
static __device__ __forceinline__ void tdm_load_tile(unsigned lds_addr, const void* gp) {
  unsigned long long ga = (unsigned long long)(uintptr_t)gp;
  u32x4 g0;
  g0[0] = 1u;                                                 // count=1, user mode
  g0[1] = lds_addr;                                           // lds_addr[31:0]
  g0[2] = (unsigned)ga;                                       // global_addr[31:0]
  g0[3] = (unsigned)((ga >> 32) & 0x01FFFFFFu) | (2u << 30);  // addr[56:32] | type=2
  i32x8 g1;
  g1[0] = (int)(3u << 16);      // wg_mask=0, data_size=3 (8B), no flags
  g1[1] = (int)(16u << 16);     // tensor_dim0[15:0]=16 units (bits 63:48)
  g1[2] = (int)(2048u << 16);   // tensor_dim0 hi=0 ; tensor_dim1[15:0]=2048
  g1[3] = (int)(16u << 16);     // tensor_dim1 hi=0 ; tile_dim0=16 units
  g1[4] = 32;                   // tile_dim1=32 rows ; tile_dim2=0
  g1[5] = 16;                   // tensor_dim0_stride[31:0]=16 units
  g1[6] = 0;                    // stride0 hi ; tensor_dim1_stride lo
  g1[7] = 0;
#if __has_builtin(__builtin_amdgcn_tensor_load_to_lds)
  i32x4 gz = {};
#if __clang_major__ >= 23
  i32x8 gz8 = {};
  __builtin_amdgcn_tensor_load_to_lds(g0, g1, gz, gz, gz8, 0);
#else
  __builtin_amdgcn_tensor_load_to_lds(g0, g1, gz, gz, 0);
#endif
#else
  asm volatile("tensor_load_to_lds %0, %1" :: "s"(g0), "s"(g1) : "memory");
#endif
}

static __device__ __forceinline__ void async_b128(unsigned lds_addr, const void* gp) {
  asm volatile("global_load_async_to_lds_b128 %0, %1, off"
               :: "v"(lds_addr), "v"(gp) : "memory");
}

// =====================================================================
// Kernel 1: QKV projection  [B,N,D] x [H,D,HS] -> bf16 [B,H,N,HS]
// Block = 8 waves sharing one (h, q/k/v); weight panel double-buffered
// into LDS via async-to-LDS. One wave -> 16x64 output tile.
// =====================================================================
__global__ __launch_bounds__(256)
void qkv_proj_kernel(const float* __restrict__ q, const float* __restrict__ k,
                     const float* __restrict__ v,
                     const float* __restrict__ wq, const float* __restrict__ wk,
                     const float* __restrict__ wv,
                     unsigned short* __restrict__ qo, unsigned short* __restrict__ ko,
                     unsigned short* __restrict__ vo)
{
  __shared__ float wlds[2][32][HS_];   // 16 KB double-buffered weight tile

  const int lane  = threadIdx.x & 31;
  const int wave  = threadIdx.x >> 5;
  const int am    = lane & 15;   // A row / B,C column within tile
  const int ahalf = lane >> 4;

  const int rowTile = blockIdx.x * 8 + wave;     // over B*N/16 = 512
  const int h       = blockIdx.y;
  const int which   = blockIdx.z;

  const float* x; const float* w; __bf16* o; float scale;
  if (which == 0)      { x = q; w = wq; o = (__bf16*)qo; scale = 0.125f; } // 1/sqrt(64)
  else if (which == 1) { x = k; w = wk; o = (__bf16*)ko; scale = 1.0f; }
  else                 { x = v; w = wv; o = (__bf16*)vo; scale = 1.0f; }

  const int b    = rowTile / (N_ / 16);
  const int row0 = (rowTile % (N_ / 16)) * 16;

  const float* arow = x + ((size_t)b * N_ + row0 + am) * D_;
  const float* wh   = w + (size_t)h * D_ * HS_;          // [D, HS]

  // staging map: 256 threads x 32B = one 32x64 f32 tile
  const int sr = threadIdx.x >> 3;         // tile row 0..31
  const int sc = (threadIdx.x & 7) * 8;    // f32 col 0,8,..,56

  auto issue_w = [&](int k0, int buf) {
    unsigned ldsa = (unsigned)(uintptr_t)&wlds[buf][sr][sc];
    const float* g = wh + (size_t)(k0 + sr) * HS_ + sc;
    async_b128(ldsa, g);
    async_b128(ldsa + 16, g + 4);
  };

  v8f acc[4] = {};
  issue_w(0, 0);

  for (int k0 = 0; k0 < D_; k0 += 32) {
    wait_async0();
    __syncthreads();
    if (k0 + 32 < D_) issue_w(k0 + 32, ((k0 >> 5) + 1) & 1);
    const int cur = (k0 >> 5) & 1;

    // A fragment: 16x32 bf16 from global (rows private to this wave)
    v16bf a;
#pragma unroll
    for (int i = 0; i < 8; ++i) {
      a[i]     = (__bf16)(arow[k0 + ahalf * 16 + i] * scale);
      a[8 + i] = (__bf16)(arow[k0 + ahalf * 16 + 8 + i] * scale);
    }
#pragma unroll
    for (int j = 0; j < 4; ++j) {
      const int col = j * 16 + am;
      v16bf bf;
#pragma unroll
      for (int i = 0; i < 16; ++i)
        bf[i] = (__bf16)wlds[cur][ahalf * 16 + i][col];
      acc[j] = __builtin_amdgcn_wmma_f32_16x16x32_bf16(
          false, a, false, bf, (short)0, acc[j], false, false);
    }
  }

  // store C tiles: N = am, M = ahalf*8 + r
  __bf16* ob = o + (((size_t)b * H_ + h) * N_ + row0) * HS_;
#pragma unroll
  for (int j = 0; j < 4; ++j) {
    const int col = j * 16 + am;
#pragma unroll
    for (int r = 0; r < 8; ++r)
      ob[(size_t)(ahalf * 8 + r) * HS_ + col] = (__bf16)acc[j][r];
  }
}

// =====================================================================
// Kernel 2: flash attention. Block = 8 waves sharing one (b,h).
// K tile: TDM tensor_load_to_lds (TENSORcnt, issued by wave 0).
// V tile: async-to-LDS copies (ASYNCcnt, all threads). Double-buffered.
// Each wave owns 16 query rows; online softmax; O accumulated in f32.
// =====================================================================
__global__ __launch_bounds__(256)
void flash_attn_kernel(const unsigned short* __restrict__ qb,
                       const unsigned short* __restrict__ kb,
                       const unsigned short* __restrict__ vb,
                       unsigned short* __restrict__ ctxo)
{
  __shared__ __bf16 ldsK[2][32][HS_];   // 8 KB  (double-buffered 32x64 K tile)
  __shared__ __bf16 ldsV[2][32][HS_];   // 8 KB
  __shared__ __bf16 plds[8][16][32];    // 8 KB  per-wave P staging (C->A relayout)

  const int lane  = threadIdx.x & 31;
  const int wave  = threadIdx.x >> 5;
  const int am    = lane & 15;
  const int ahalf = lane >> 4;

  const int nt = blockIdx.x * 8 + wave;   // query tile 0..N/16-1
  const int h  = blockIdx.y;
  const int b  = blockIdx.z;

  const __bf16* Q = (const __bf16*)qb + (((size_t)b * H_ + h) * N_) * HS_;
  const __bf16* K = (const __bf16*)kb + (((size_t)b * H_ + h) * M_) * HS_;
  const __bf16* V = (const __bf16*)vb + (((size_t)b * H_ + h) * M_) * HS_;

  // cooperative V staging: 256 threads x 16B = one 32x64 bf16 tile
  const int sr = threadIdx.x >> 3;         // tile row 0..31
  const int sc = (threadIdx.x & 7) * 8;    // element col 0,8,..,56

  auto issue_v = [&](int m0, int buf) {
    unsigned ldsa = (unsigned)(uintptr_t)&ldsV[buf][sr][sc];
    async_b128(ldsa, V + (size_t)(m0 + sr) * HS_ + sc);
  };
  auto issue_k = [&](int m0, int buf) {
    tdm_load_tile((unsigned)(uintptr_t)&ldsK[buf][0][0],
                  K + (size_t)m0 * HS_);
  };

  // Q fragments (loop invariant): 16x64 -> two 16x32 A fragments
  const __bf16* qrow = Q + (size_t)(nt * 16 + am) * HS_;
  v16bf aq[2];
#pragma unroll
  for (int kk = 0; kk < 2; ++kk)
#pragma unroll
    for (int i = 0; i < 8; ++i) {
      aq[kk][i]     = qrow[kk * 32 + ahalf * 16 + i];
      aq[kk][8 + i] = qrow[kk * 32 + ahalf * 16 + 8 + i];
    }

  v8f   accO[4] = {};
  float mrow[8], lrow[8];
#pragma unroll
  for (int r = 0; r < 8; ++r) { mrow[r] = -1e30f; lrow[r] = 0.0f; }

  if (wave == 0) issue_k(0, 0);
  issue_v(0, 0);

  for (int it = 0; it < M_ / 32; ++it) {
    // drain own DMA (per-wave counters); barrier => tile `it` resident for
    // all waves, and every wave is done reading the next issue's target buf
    wait_async0();
    if (wave == 0) wait_tensor0();
    __syncthreads();
    if (it + 1 < M_ / 32) {
      if (wave == 0) issue_k((it + 1) * 32, (it + 1) & 1);
      issue_v((it + 1) * 32, (it + 1) & 1);
    }
    const int cur = it & 1;

    // ---- S = Q K^T : 16x32 tile (K rows contiguous in LDS -> b128 reads)
    v8f accS[2] = {};
#pragma unroll
    for (int jt = 0; jt < 2; ++jt) {
#pragma unroll
      for (int kk = 0; kk < 2; ++kk) {
        const __bf16* kp = &ldsK[cur][jt * 16 + am][kk * 32 + ahalf * 16];
        v16bf bf;
#pragma unroll
        for (int i = 0; i < 16; ++i) bf[i] = kp[i];
        accS[jt] = __builtin_amdgcn_wmma_f32_16x16x32_bf16(
            false, aq[kk], false, bf, (short)0, accS[jt], false, false);
      }
    }

    // ---- online softmax (row r lives in VGPR r across this 16-lane half)
    float alpha[8];
#pragma unroll
    for (int r = 0; r < 8; ++r) {
      float mx   = half_max(fmaxf(accS[0][r], accS[1][r]));
      float mnew = fmaxf(mrow[r], mx);
      alpha[r]   = __expf(mrow[r] - mnew);
      float p0   = __expf(accS[0][r] - mnew);
      float p1   = __expf(accS[1][r] - mnew);
      accS[0][r] = p0;
      accS[1][r] = p1;
      lrow[r] = lrow[r] * alpha[r] + half_sum(p0 + p1);
      mrow[r] = mnew;
    }
#pragma unroll
    for (int j = 0; j < 4; ++j)
#pragma unroll
      for (int r = 0; r < 8; ++r) accO[j][r] *= alpha[r];

    // ---- P: C-layout -> LDS row-major -> A fragment (same-wave DS, in order)
#pragma unroll
    for (int jt = 0; jt < 2; ++jt)
#pragma unroll
      for (int r = 0; r < 8; ++r)
        plds[wave][ahalf * 8 + r][jt * 16 + am] = (__bf16)accS[jt][r];

    v16bf ap;
#pragma unroll
    for (int i = 0; i < 8; ++i) {
      ap[i]     = plds[wave][am][ahalf * 16 + i];
      ap[8 + i] = plds[wave][am][ahalf * 16 + 8 + i];
    }

    // ---- O += P V (V fragment gathered from LDS)
#pragma unroll
    for (int j = 0; j < 4; ++j) {
      const int col = j * 16 + am;
      v16bf bf;
#pragma unroll
      for (int i = 0; i < 16; ++i) bf[i] = ldsV[cur][ahalf * 16 + i][col];
      accO[j] = __builtin_amdgcn_wmma_f32_16x16x32_bf16(
          false, ap, false, bf, (short)0, accO[j], false, false);
    }
  }

  // ---- normalize and store ctx (bf16)
  __bf16* ob = (__bf16*)ctxo + (((size_t)b * H_ + h) * N_ + nt * 16) * HS_;
#pragma unroll
  for (int r = 0; r < 8; ++r) {
    const float inv = 1.0f / lrow[r];
#pragma unroll
    for (int j = 0; j < 4; ++j)
      ob[(size_t)(ahalf * 8 + r) * HS_ + (j * 16 + am)] = (__bf16)(accO[j][r] * inv);
  }
}

// =====================================================================
// Kernel 3: output projection  ctx[B,H,N,HS] x proj[H,HS,OUT] -> [B,N,OUT]
// contraction over fused (h,hs) = 1024; weight panel (shared by block)
// double-buffered into LDS via async-to-LDS. One wave -> 16x64 tile.
// =====================================================================
__global__ __launch_bounds__(256)
void out_proj_kernel(const unsigned short* __restrict__ ctxi,
                     const float* __restrict__ wproj,
                     const float* __restrict__ bias,
                     float* __restrict__ out)
{
  __shared__ float wlds[2][32][64];   // 16 KB double-buffered weight tile

  const int lane  = threadIdx.x & 31;
  const int wave  = threadIdx.x >> 5;
  const int am    = lane & 15;
  const int ahalf = lane >> 4;

  const int rowTile = blockIdx.x * 8 + wave;   // over B*N/16 = 512
  const int out0    = blockIdx.y * 64;

  const int b    = rowTile / (N_ / 16);
  const int row0 = (rowTile % (N_ / 16)) * 16;

  const __bf16* ctx = (const __bf16*)ctxi;

  const int sr = threadIdx.x >> 3;
  const int sc = (threadIdx.x & 7) * 8;

  auto issue_w = [&](int k0, int buf) {
    unsigned ldsa = (unsigned)(uintptr_t)&wlds[buf][sr][sc];
    const float* g = wproj + (size_t)(k0 + sr) * OUT_ + out0 + sc;
    async_b128(ldsa, g);
    async_b128(ldsa + 16, g + 4);
  };

  v8f acc[4] = {};
  issue_w(0, 0);

  for (int k0 = 0; k0 < H_ * HS_; k0 += 32) {
    wait_async0();
    __syncthreads();
    if (k0 + 32 < H_ * HS_) issue_w(k0 + 32, ((k0 >> 5) + 1) & 1);
    const int cur = (k0 >> 5) & 1;

    const int h   = k0 / HS_;
    const int hs0 = k0 % HS_;      // 0 or 32, stays within one head block
    const __bf16* crow = ctx + (((size_t)b * H_ + h) * N_ + row0 + am) * HS_ + hs0;
    v16bf a;
#pragma unroll
    for (int i = 0; i < 8; ++i) {
      a[i]     = crow[ahalf * 16 + i];
      a[8 + i] = crow[ahalf * 16 + 8 + i];
    }
#pragma unroll
    for (int j = 0; j < 4; ++j) {
      const int col = j * 16 + am;
      v16bf bf;
#pragma unroll
      for (int i = 0; i < 16; ++i)
        bf[i] = (__bf16)wlds[cur][ahalf * 16 + i][col];
      acc[j] = __builtin_amdgcn_wmma_f32_16x16x32_bf16(
          false, a, false, bf, (short)0, acc[j], false, false);
    }
  }

  float* ob = out + ((size_t)b * N_ + row0) * OUT_;
#pragma unroll
  for (int j = 0; j < 4; ++j) {
    const int col = out0 + j * 16 + am;
    const float bv = bias[col];
#pragma unroll
    for (int r = 0; r < 8; ++r)
      ob[(size_t)(ahalf * 8 + r) * OUT_ + col] = acc[j][r] + bv;
  }
}

// =====================================================================
extern "C" void kernel_launch(void* const* d_in, const int* in_sizes, int n_in,
                              void* d_out, int out_size, void* d_ws, size_t ws_size,
                              hipStream_t stream) {
  const float* query = (const float*)d_in[0];
  const float* key   = (const float*)d_in[1];
  const float* value = (const float*)d_in[2];
  const float* wq    = (const float*)d_in[3];
  const float* wk    = (const float*)d_in[4];
  const float* wv    = (const float*)d_in[5];
  const float* wproj = (const float*)d_in[6];
  const float* bias  = (const float*)d_in[7];
  float* out = (float*)d_out;

  // workspace: 4 bf16 tensors of [B,H,N,HS] = 16 MB each (64 MB total)
  const size_t bytesPer = (size_t)B_ * H_ * N_ * HS_ * sizeof(unsigned short);
  unsigned short* qb  = (unsigned short*)d_ws;
  unsigned short* kb  = (unsigned short*)((char*)d_ws + bytesPer);
  unsigned short* vb  = (unsigned short*)((char*)d_ws + 2 * bytesPer);
  unsigned short* cb  = (unsigned short*)((char*)d_ws + 3 * bytesPer);

  dim3 blk(256);
  qkv_proj_kernel<<<dim3((B_ * N_ / 16) / 8, H_, 3), blk, 0, stream>>>(
      query, key, value, wq, wk, wv, qb, kb, vb);
  flash_attn_kernel<<<dim3((N_ / 16) / 8, H_, B_), blk, 0, stream>>>(qb, kb, vb, cb);
  out_proj_kernel<<<dim3((B_ * N_ / 16) / 8, OUT_ / 64), blk, 0, stream>>>(
      cb, wproj, bias, out);
}
